// ScaleDotProduct_55602646614387
// MI455X (gfx1250) — matrix operand order repaired and checked by reading, not compiled, not run on hardware
//
#include <hip/hip_runtime.h>
#include <hip/hip_bf16.h>

// Problem shape (fixed by the reference)
constexpr int Bb = 4, Hh = 16, Ss = 2048, Dd = 64;
constexpr int KT = 64;                 // keys per staged tile
constexpr int NJ = Ss / KT;            // 32 tiles

typedef __attribute__((ext_vector_type(16))) __bf16 v16bf;
typedef __attribute__((ext_vector_type(2)))  __bf16 v2bf;
typedef __attribute__((ext_vector_type(2)))  float  v2f;
typedef __attribute__((ext_vector_type(8)))  float  v8f;

union Frag16 {                         // 32 bytes: one bf16 A/B fragment
  uint4    u[2];                       // two 16B chunks (LDS b128 loads)
  unsigned w[8];                       // scalar word view (register packing)
  v16bf    v;
};

__device__ __forceinline__ unsigned short bf16_bits(float f) {
  __bf16 h = (__bf16)f;                            // native v_cvt path, RTE
  return __builtin_bit_cast(unsigned short, h);
}
// packed pair convert -> single v_cvt_pk_bf16_f32 (two-operand form)
#if __has_builtin(__builtin_amdgcn_cvt_pk_bf16_f32)
__device__ __forceinline__ unsigned pk_bf16(float lo, float hi) {
  v2bf r = __builtin_amdgcn_cvt_pk_bf16_f32(lo, hi);
  return __builtin_bit_cast(unsigned, r);
}
#else
__device__ __forceinline__ unsigned pk_bf16(float lo, float hi) {
  v2f  f = {lo, hi};
  v2bf r = __builtin_convertvector(f, v2bf);
  return __builtin_bit_cast(unsigned, r);
}
#endif

// LDS strides (bf16 elements); all fragment sub-reads stay 16B aligned
#define SK_STRIDE  72   // 64 rows (key) x (64 d + pad);   row byte stride 144
#define SVT_STRIDE 72   // 64 rows (d)   x (64 key + pad); row byte stride 144

__global__ __launch_bounds__(256)
void ScaleDotProduct_fa_kernel(const float* __restrict__ Qg,
                               const float* __restrict__ Kg,
                               const float* __restrict__ Vg,
                               const float* __restrict__ scale_p,
                               float* __restrict__ Og) {
  // double-buffered tiles: stage (j+1) while computing (j)
  __shared__ unsigned short sK [2][KT * SK_STRIDE];    // K tile, row-major (key, d)
  __shared__ unsigned short sVt[2][Dd * SVT_STRIDE];   // V tile, transposed (d, key)

  const int tid  = threadIdx.x;
  const int lane = tid & 31;
  const int wave = tid >> 5;
  const int g    = lane >> 4;        // 16-lane half
  const int n    = lane & 15;        // column index inside fragment
  const int g8   = g * 8;

  const int bh = blockIdx.y;                       // 0..B*H-1
  const int q0 = (blockIdx.x * 8 + wave) * 16;     // this wave's query rows
  const size_t base = (size_t)bh * Ss * Dd;

  // softmax scale folded with log2(e): exp(x) == exp2(x * log2e)
  const float scale = scale_p[0] * 1.4426950408889634f;

  // ---- cooperative tile stage: K row-major; V transposed (natural key order)
  auto stage = [&](int jt, int buf) {
    const int kk = tid >> 2;            // key row 0..63
    const int d0 = (tid & 3) * 16;      // 16 contiguous d per thread
    const float* kp = Kg + base + (size_t)(jt * KT + kk) * Dd + d0;
    const float* vp = Vg + base + (size_t)(jt * KT + kk) * Dd + d0;
    float4 k0 = *(const float4*)(kp);
    float4 k1 = *(const float4*)(kp + 4);
    float4 k2 = *(const float4*)(kp + 8);
    float4 k3 = *(const float4*)(kp + 12);
    float4 v0 = *(const float4*)(vp);
    float4 v1 = *(const float4*)(vp + 4);
    float4 v2 = *(const float4*)(vp + 8);
    float4 v3 = *(const float4*)(vp + 12);
    if (jt + 1 < NJ) {                  // global_prefetch_b8 for the tile after
      __builtin_prefetch(kp + KT * Dd, 0, 3);
      __builtin_prefetch(vp + KT * Dd, 0, 3);
    }
    uint4 pa, pb;
    pa.x = pk_bf16(k0.x, k0.y); pa.y = pk_bf16(k0.z, k0.w);
    pa.z = pk_bf16(k1.x, k1.y); pa.w = pk_bf16(k1.z, k1.w);
    pb.x = pk_bf16(k2.x, k2.y); pb.y = pk_bf16(k2.z, k2.w);
    pb.z = pk_bf16(k3.x, k3.y); pb.w = pk_bf16(k3.z, k3.w);
    unsigned short* skb = &sK[buf][0];
    *(uint4*)&skb[kk * SK_STRIDE + d0]     = pa;   // ds_store_b128 x2
    *(uint4*)&skb[kk * SK_STRIDE + d0 + 8] = pb;

    unsigned short* svb = &sVt[buf][0];
    float vv[16] = {v0.x,v0.y,v0.z,v0.w, v1.x,v1.y,v1.z,v1.w,
                    v2.x,v2.y,v2.z,v2.w, v3.x,v3.y,v3.z,v3.w};
#pragma unroll
    for (int t = 0; t < 16; ++t)
      svb[(d0 + t) * SVT_STRIDE + kk] = bf16_bits(vv[t]);
  };

  // ---- Q as B-side fragments (Q^T): lane = query n, element e -> d = 8g+e (e<8), 16+8g+(e-8) (e>=8)
  const float* qrow = Qg + base + (size_t)(q0 + n) * Dd;
  Frag16 bQ0, bQ1;
#pragma unroll
  for (int i = 0; i < 8; ++i) {
    bQ0.w[i] = pk_bf16(qrow[     8*g + 2*i + 2*(i&4)] * scale, qrow[     8*g + 2*i + 2*(i&4)+1] * scale);
    bQ1.w[i] = pk_bf16(qrow[32 + 8*g + 2*i + 2*(i&4)] * scale, qrow[32 + 8*g + 2*i + 2*(i&4)+1] * scale);
  }

  // ---- state: acc in C layout (lane: rows 8g+r, col d=n); m/l per query-lane
  v8f acc[4] = {{0.f,0.f,0.f,0.f,0.f,0.f,0.f,0.f},
                {0.f,0.f,0.f,0.f,0.f,0.f,0.f,0.f},
                {0.f,0.f,0.f,0.f,0.f,0.f,0.f,0.f},
                {0.f,0.f,0.f,0.f,0.f,0.f,0.f,0.f}};
  float mrow = -1.0e30f;   // running max for query n (this lane)
  float lrow = 0.0f;       // running denominator for query n

  stage(0, 0);                       // prologue: fill buffer 0

  for (int j = 0; j < NJ; ++j) {
    __syncthreads();   // buf[j&1] staged; prior reads of buf[(j+1)&1] retired
    const int cur = j & 1;
    stage((j + 1) & (NJ - 1), cur ^ 1);   // overlap next tile with this compute

    // ---- transposed scores: sT[t] = K_tile(t) @ Q^T  (A = K rows from LDS)
    const unsigned short* skb = &sK[cur][0];
    v8f sT[4];
#pragma unroll
    for (int t = 0; t < 4; ++t) {
      const unsigned short* kr = &skb[(t * 16 + n) * SK_STRIDE];  // A row = key
      Frag16 a0, a1;
      a0.u[0] = *(const uint4*)(kr + g8);            // d = 8g+e
      a0.u[1] = *(const uint4*)(kr + 16 + g8);       // d = 16+8g+e
      a1.u[0] = *(const uint4*)(kr + 32 + g8);       // d-half 1
      a1.u[1] = *(const uint4*)(kr + 48 + g8);
      v8f c = {0.f,0.f,0.f,0.f,0.f,0.f,0.f,0.f};
      c = __builtin_amdgcn_wmma_f32_16x16x32_bf16(false, a0.v, false, bQ0.v,
                                                  (short)0, c, false, false);
      c = __builtin_amdgcn_wmma_f32_16x16x32_bf16(false, a1.v, false, bQ1.v,
                                                  (short)0, c, false, false);
      sT[t] = c;   // lane: query n, keys t*16 + 8g + r
    }

    // ---- online softmax: balanced in-lane trees + one cross-half exchange
    float tm[4];
#pragma unroll
    for (int t = 0; t < 4; ++t) {
      float a = fmaxf(fmaxf(sT[t][0], sT[t][1]), fmaxf(sT[t][2], sT[t][3]));
      float b = fmaxf(fmaxf(sT[t][4], sT[t][5]), fmaxf(sT[t][6], sT[t][7]));
      tm[t] = fmaxf(a, b);
    }
    float m1 = fmaxf(fmaxf(tm[0], tm[1]), fmaxf(tm[2], tm[3]));
    m1 = fmaxf(m1, __shfl_xor(m1, 16, 32));
    const float mnew  = fmaxf(mrow, m1);
    const float alpha = __builtin_exp2f(mrow - mnew);
    mrow = mnew;

    float pr[4][8];
    float ts[4];
#pragma unroll
    for (int t = 0; t < 4; ++t) {
#pragma unroll
      for (int r = 0; r < 8; ++r)
        pr[t][r] = __builtin_exp2f(sT[t][r] - mnew);
      float a = (pr[t][0] + pr[t][1]) + (pr[t][2] + pr[t][3]);
      float b = (pr[t][4] + pr[t][5]) + (pr[t][6] + pr[t][7]);
      ts[t] = a + b;
    }
    float ps = (ts[0] + ts[1]) + (ts[2] + ts[3]);
    ps += __shfl_xor(ps, 16, 32);
    lrow = lrow * alpha + ps;

    // broadcast alpha from query-lanes to acc-row layout (rows 8g+r)
    float alr[8];
#pragma unroll
    for (int r = 0; r < 8; ++r) alr[r] = __shfl(alpha, g8 + r, 32);
#pragma unroll
    for (int cc = 0; cc < 4; ++cc)
#pragma unroll
      for (int r = 0; r < 8; ++r) acc[cc][r] *= alr[r];

    // ---- P fragments built IN-REGISTER: Sᵀ C-layout == P A-layout (no LDS!)
    Frag16 ap[2];
#pragma unroll
    for (int kh = 0; kh < 2; ++kh)
#pragma unroll
      for (int i = 0; i < 4; ++i) {
        ap[kh].w[i]     = pk_bf16(pr[2*kh][2*i],     pr[2*kh][2*i+1]);     // K=8g+e
        ap[kh].w[4 + i] = pk_bf16(pr[2*kh+1][2*i],   pr[2*kh+1][2*i+1]);   // K=16+8g+e
      }

    // ---- O += P @ V : 4 d-chunks x 2 key-halves = 8 WMMAs
    const unsigned short* svb = &sVt[cur][0];
#pragma unroll
    for (int cc = 0; cc < 4; ++cc) {
      const unsigned short* vr = &svb[(cc * 16 + n) * SVT_STRIDE]; // B col = d
#pragma unroll
      for (int kh = 0; kh < 2; ++kh) {
        Frag16 bv;
        bv.u[0] = *(const uint4*)(vr + kh * 32 + g8);              // K(key)=8g+e
        bv.u[1] = *(const uint4*)(vr + kh * 32 + 16 + g8);
        acc[cc] = __builtin_amdgcn_wmma_f32_16x16x32_bf16(false, ap[kh].v, false,
                                                          bv.v, (short)0, acc[cc],
                                                          false, false);
      }
    }
  }

  // ---- epilogue: broadcast 1/l to acc-row layout, store fp32 output
  const float inv = 1.0f / lrow;
  float invr[8];
#pragma unroll
  for (int r = 0; r < 8; ++r) invr[r] = __shfl(inv, g8 + r, 32);
#pragma unroll
  for (int r = 0; r < 8; ++r) {
    const int m = g8 + r;
    float* op = Og + base + (size_t)(q0 + m) * Dd + n;
#pragma unroll
    for (int cc = 0; cc < 4; ++cc) op[cc * 16] = acc[cc][r] * invr[r];
  }
}

extern "C" void kernel_launch(void* const* d_in, const int* in_sizes, int n_in,
                              void* d_out, int out_size, void* d_ws, size_t ws_size,
                              hipStream_t stream) {
  const float* Q  = (const float*)d_in[0];
  const float* K  = (const float*)d_in[1];
  const float* V  = (const float*)d_in[2];
  const float* sc = (const float*)d_in[3];
  float* O = (float*)d_out;
  dim3 grid(Ss / 128, Bb * Hh);   // (16, 64): 8 query tiles of 16 per block
  ScaleDotProduct_fa_kernel<<<grid, 256, 0, stream>>>(Q, K, V, sc, O);
}
